// FLASH_40458591928592
// MI455X (gfx1250) — matrix operand order repaired and checked by reading, not compiled, not run on hardware
//
#include <hip/hip_runtime.h>
#include <hip/hip_bf16.h>

typedef _Float16 h16;
typedef __attribute__((ext_vector_type(16))) _Float16 v16h;
typedef __attribute__((ext_vector_type(4)))  _Float16 v4h;
typedef __attribute__((ext_vector_type(8)))  float    v8f;

#define BM 128
#define BN 128
#define BKK 32
#define PADK 8
#define THREADS 256

#define G_SZ   256
#define N_TOK  8192
#define DIM    1024
#define HID    2048
#define DQK    128
#define NGRP   (N_TOK / G_SZ)
#define KTAPS  32   // conv_w masked to taps 0..31

__device__ __forceinline__ v8f wmma16(v16h a, v16h b, v8f c) {
  // D = A(16x32 f16) * B(32x16 f16) + C(16x16 f32)
  return __builtin_amdgcn_wmma_f32_16x16x32_f16(false, a, false, b, (short)0, c, false, false);
}

// CDNA5 async global->LDS copy (ASYNCcnt-tracked), 16 bytes per lane.
__device__ __forceinline__ void async_g2l_b128(void* lds, const void* gsrc) {
  asm volatile("global_load_async_to_lds_b128 %0, %1, off"
               :: "v"((unsigned)(unsigned long long)lds), "v"(gsrc)
               : "memory");
}
__device__ __forceinline__ void wait_async0() {
  asm volatile("s_wait_asynccnt 0x0" ::: "memory");
}

// Epilogue codes
#define EPI_SILU_F16   0  // Ch = silu(acc + bias[n])            (f16; CTRANS => store [n][m])
#define EPI_ATTN_F16   1  // Ch = mask(relu(acc*scale)^2)        (f16, causal in-group)
#define EPI_SCALE_F32  2  // Cf = acc * scale
#define EPI_ADD_F32    3  // Cf = acc + add (ADDTRANS => add[n][m])
#define EPI_ACC_F32    4  // Cf += acc
#define EPI_FINAL_F32  5  // Cf = acc + bias[n] + add[m][n]      (b_out + residual x)

struct GemmP {
  const h16*  A;   long lda;  long strideA;               // AMODE 0: MxK row-major
  const h16*  B;   long ldb;  long strideB;               // B always NxK row-major
  const float* A2; long lda2; int arow_mod;               // AMODE 2: A=(f16)(A[m][k]*A2[m%mod][k])
  int   K;
  float scale;
  float*       Cf;   h16* Ch;  long ldc;   long strideC;
  const float* bias;
  const float* add;  long ldadd; long strideAdd;
};

template<int AMODE, int EPI, bool CTRANS, bool ADDTRANS>
__global__ __launch_bounds__(THREADS) void flash_gemm(GemmP p) {
  __shared__ h16 As[2][BM][BKK + PADK];   // row stride 80B (16B-aligned)
  __shared__ h16 Bs[2][BN][BKK + PADK];   // Bs[n][k]

  const int tid  = threadIdx.x;
  const int lane = tid & 31;
  const int wave = tid >> 5;
  const int wm   = wave >> 1;      // 0..3 -> M offset wm*32
  const int wn   = wave & 1;       // 0..1 -> N offset wn*64
  const int r    = lane & 15;
  const int hh   = lane >> 4;

  const long tileN = (long)blockIdx.x * BN;
  const long tileM = (long)blockIdx.y * BM;
  const int  batch = blockIdx.z;

  const h16* Ab = p.A + (long)batch * p.strideA;
  const h16* Bb = p.B + (long)batch * p.strideB;

  auto stage = [&](int kt, int buf) {
    if (AMODE == 0) {
#pragma unroll
      for (int c = tid; c < BM * BKK / 8; c += THREADS) {   // 512 x 16B chunks
        int row = c >> 2, col = (c & 3) * 8;
        async_g2l_b128(&As[buf][row][col], Ab + (tileM + row) * p.lda + kt + col);
      }
    } else {  // gated: A = gate(f16) * outpre(f32) -> f16
#pragma unroll
      for (int c = tid; c < BM * BKK / 4; c += THREADS) {
        int row = c >> 3, col = (c & 7) * 4;
        long gm = tileM + row;
        v4h g4 = *(const v4h*)(Ab + gm * p.lda + kt + col);
        const float* o = p.A2 + (gm % p.arow_mod) * p.lda2 + kt + col;
        v4h v;
#pragma unroll
        for (int e = 0; e < 4; e++) v[e] = (h16)((float)g4[e] * o[e]);
        *(v4h*)&As[buf][row][col] = v;
      }
    }
#pragma unroll
    for (int c = tid; c < BN * BKK / 8; c += THREADS) {
      int n = c >> 2, col = (c & 3) * 8;
      async_g2l_b128(&Bs[buf][n][col], Bb + (tileN + n) * p.ldb + kt + col);
    }
  };

  v8f acc[2][4];
#pragma unroll
  for (int i = 0; i < 2; i++)
#pragma unroll
    for (int j = 0; j < 4; j++)
#pragma unroll
      for (int e = 0; e < 8; e++) acc[i][j][e] = 0.f;

  stage(0, 0);
  wait_async0();
  __syncthreads();

  int buf = 0;
  for (int kt = 0; kt < p.K; kt += BKK) {
    const int nbuf = buf ^ 1;
    if (kt + BKK < p.K) stage(kt + BKK, nbuf);   // prefetch next tile (async)

    // ---- fragments (ISA 7.12.2 layouts) ----
    v16h a[2], b[4];
#pragma unroll
    for (int i = 0; i < 2; i++) {
      const h16* row = &As[buf][wm * 32 + i * 16 + r][0];
#pragma unroll
      for (int e = 0; e < 8; e++) { a[i][e] = row[8 * hh + e]; a[i][8 + e] = row[16 + 8 * hh + e]; }
    }
#pragma unroll
    for (int j = 0; j < 4; j++) {
      const h16* row = &Bs[buf][wn * 64 + j * 16 + r][0];
#pragma unroll
      for (int e = 0; e < 8; e++) { b[j][e] = row[8 * hh + e]; b[j][8 + e] = row[16 + 8 * hh + e]; }
    }
#pragma unroll
    for (int i = 0; i < 2; i++)
#pragma unroll
      for (int j = 0; j < 4; j++)
        acc[i][j] = wmma16(a[i], b[j], acc[i][j]);

    wait_async0();
    __syncthreads();
    buf = nbuf;
  }

  // ---- epilogue: D element (m = rr + 8*hh, n = r) per vgpr rr ----
#pragma unroll
  for (int i = 0; i < 2; i++) {
#pragma unroll
    for (int j = 0; j < 4; j++) {
#pragma unroll
      for (int rr = 0; rr < 8; rr++) {
        long gm = tileM + wm * 32 + i * 16 + rr + 8 * hh;
        long gn = tileN + wn * 64 + j * 16 + r;
        long off = (long)batch * p.strideC + gm * p.ldc + gn;
        float val = acc[i][j][rr];
        if (EPI == EPI_SILU_F16) {
          float s = val + p.bias[gn];
          s = s / (1.f + __expf(-s));
          if (CTRANS) p.Ch[(long)batch * p.strideC + gn * p.ldc + gm] = (h16)s;
          else        p.Ch[off] = (h16)s;
        } else if (EPI == EPI_ATTN_F16) {
          float s = val * p.scale;
          s = (gn > gm) ? 0.f : (s > 0.f ? s * s : 0.f);
          p.Ch[off] = (h16)s;
        } else if (EPI == EPI_SCALE_F32) {
          p.Cf[off] = val * p.scale;
        } else if (EPI == EPI_ADD_F32) {
          long aoff = ADDTRANS
            ? (long)batch * p.strideAdd + gn * p.ldadd + gm
            : (long)batch * p.strideAdd + gm * p.ldadd + gn;
          p.Cf[off] = val + p.add[aoff];
        } else if (EPI == EPI_ACC_F32) {
          p.Cf[off] += val;
        } else { // EPI_FINAL_F32
          p.Cf[off] = val + p.bias[gn] + p.add[gm * p.ldadd + gn];
        }
      }
    }
  }
}

__global__ void flash_cvt(const float* __restrict__ in, h16* __restrict__ out, long n) {
  long i = (long)blockIdx.x * blockDim.x + threadIdx.x;
  if (i < n) out[i] = (h16)in[i];
}

// f32 (rows x cols) -> f16 transposed (cols x rows)
__global__ void flash_cvt_t(const float* __restrict__ in, h16* __restrict__ out,
                            long rows, long cols) {
  long i = (long)blockIdx.x * blockDim.x + threadIdx.x;
  if (i >= rows * cols) return;
  long rr = i / cols, cc = i % cols;
  out[cc * rows + rr] = (h16)in[i];
}

// f16 (rows x cols) -> f16 transposed (cols x rows)
__global__ void flash_tr_h(const h16* __restrict__ in, h16* __restrict__ out,
                           long rows, long cols) {
  long i = (long)blockIdx.x * blockDim.x + threadIdx.x;
  if (i >= rows * cols) return;
  long rr = i / cols, cc = i % cols;
  out[cc * rows + rr] = in[i];
}

// exclusive prefix over groups: kvpreT[g][e][d] = sum_{g'<g} kv[g'][d][e]
__global__ void flash_cumsum(const float* __restrict__ kv, h16* __restrict__ kvpreT, long per) {
  long i = (long)blockIdx.x * blockDim.x + threadIdx.x;
  if (i >= per) return;
  long d = i / HID, e = i % HID;
  float s = 0.f;
  for (int g = 0; g < NGRP; g++) {
    kvpreT[(long)g * per + e * DQK + d] = (h16)s;
    s += kv[(long)g * per + i];
  }
}

// causal depthwise conv, 32 taps, transposed layout: convT[e][n] = sum_j w[j] * v0T[e][n + j - 31]
__global__ void flash_conv(const h16* __restrict__ v0t, const float* __restrict__ w,
                           float* __restrict__ outt) {
  long t = (long)blockIdx.x * blockDim.x + threadIdx.x;
  if (t >= (long)HID * (N_TOK / 8)) return;
  int e  = (int)(t >> 10);            // N_TOK/8 == 1024
  int n0 = (int)(t & 1023) * 8;
  const h16* rowp = v0t + (long)e * N_TOK;
  float win[39];
#pragma unroll
  for (int i = 0; i < 39; i++) {
    int m = n0 - (KTAPS - 1) + i;
    win[i] = (m >= 0) ? (float)rowp[m] : 0.f;
  }
  float wr[KTAPS];
#pragma unroll
  for (int j = 0; j < KTAPS; j++) wr[j] = w[j];
#pragma unroll
  for (int tt = 0; tt < 8; tt++) {
    float a = 0.f;
#pragma unroll
    for (int j = 0; j < KTAPS; j++) a += wr[j] * win[tt + j];
    outt[(long)e * N_TOK + n0 + tt] = a;
  }
}

#define LAUNCH_GEMM(AM, EP, CT, AT, P, M, N, NB)                                  \
  do { dim3 _g((N) / BN, (M) / BM, (NB));                                         \
       flash_gemm<AM, EP, CT, AT><<<_g, THREADS, 0, stream>>>(P); } while (0)

extern "C" void kernel_launch(void* const* d_in, const int* in_sizes, int n_in,
                              void* d_out, int out_size, void* d_ws, size_t ws_size,
                              hipStream_t stream) {
  const float* x      = (const float*)d_in[0];   // (4, 8192, 1024)
  const float* W_h    = (const float*)d_in[1];   // (1024, 4096)
  const float* b_h    = (const float*)d_in[2];   // (4096)
  const float* W_qk   = (const float*)d_in[3];   // (1024, 128)
  const float* b_qk   = (const float*)d_in[4];   // (128)
  const float* W_out  = (const float*)d_in[5];   // (2048, 1024)
  const float* b_out  = (const float*)d_in[6];   // (1024)
  const float* conv_w = (const float*)d_in[7];   // (63), taps 0..31 live
  float* out = (float*)d_out;                    // (4, 8192, 1024)

  char* w = (char*)d_ws;
  auto alloc = [&](size_t bytes) { char* p = w; w += (bytes + 255) & ~(size_t)255; return p; };

  h16*   xh      = (h16*)  alloc((size_t)4 * N_TOK * DIM * 2);
  h16*   WhT     = (h16*)  alloc((size_t)2 * HID * DIM * 2);   // (4096 x 1024)
  h16*   WqkT    = (h16*)  alloc((size_t)DQK * DIM * 2);       // (128 x 1024)
  h16*   WoutT   = (h16*)  alloc((size_t)DIM * HID * 2);       // (1024 x 2048)
  h16*   qk_h    = (h16*)  alloc((size_t)4 * N_TOK * DQK * 2); // token-major, 4 roles
  h16*   linkT   = (h16*)  alloc((size_t)DQK * N_TOK * 2);     // lin_k^T (128 x 8192)
  h16*   v0T     = (h16*)  alloc((size_t)HID * N_TOK * 2);     // (2048 x 8192)
  h16*   gate_h  = (h16*)  alloc((size_t)4 * N_TOK * HID * 2);
  h16*   attn_h  = (h16*)  alloc((size_t)NGRP * G_SZ * G_SZ * 2);
  float* kv_f    = (float*)alloc((size_t)NGRP * DQK * HID * 4);
  h16*   kvpreT  = (h16*)  alloc((size_t)NGRP * HID * DQK * 2); // per-group (2048 x 128)
  float* convT_f = (float*)alloc((size_t)HID * N_TOK * 4);      // (2048 x 8192)
  float* outpre  = (float*)alloc((size_t)N_TOK * HID * 4);      // (8192 x 2048)

  auto cvt = [&](const float* src, h16* dst, long n) {
    flash_cvt<<<dim3((unsigned)((n + 255) / 256)), 256, 0, stream>>>(src, dst, n);
  };
  auto cvtT = [&](const float* src, h16* dst, long rows, long cols) {
    long n = rows * cols;
    flash_cvt_t<<<dim3((unsigned)((n + 255) / 256)), 256, 0, stream>>>(src, dst, rows, cols);
  };
  cvt (x,     xh,    (long)4 * N_TOK * DIM);
  cvtT(W_h,   WhT,   DIM, 2 * HID);     // -> (4096 x 1024)
  cvtT(W_qk,  WqkT,  DIM, DQK);         // -> (128 x 1024)
  cvtT(W_out, WoutT, HID, DIM);         // -> (1024 x 2048)

  GemmP p{};

  // 1) qk = silu(x @ W_qk + b_qk)  : M=32768, N=128, K=1024
  p = GemmP{};
  p.A = xh; p.lda = DIM;
  p.B = WqkT; p.ldb = DIM;
  p.K = DIM; p.bias = b_qk;
  p.Ch = qk_h; p.ldc = DQK;
  LAUNCH_GEMM(0, EPI_SILU_F16, false, false, p, 4 * N_TOK, DQK, 1);

  // 2) v0T = silu(x0 @ W_h[:, :HID] + b_h[:HID])^T : M=8192, N=2048, K=1024
  p = GemmP{};
  p.A = xh; p.lda = DIM;
  p.B = WhT; p.ldb = DIM;
  p.K = DIM; p.bias = b_h;
  p.Ch = v0T; p.ldc = N_TOK;   // transposed store: [n][m]
  LAUNCH_GEMM(0, EPI_SILU_F16, true, false, p, N_TOK, HID, 1);

  // 3) gate_c = silu(x_c @ W_h[:, HID:] + b_h[HID:]) : M=32768, N=2048, K=1024
  p = GemmP{};
  p.A = xh; p.lda = DIM;
  p.B = WhT + (long)HID * DIM; p.ldb = DIM;
  p.K = DIM; p.bias = b_h + HID;
  p.Ch = gate_h; p.ldc = HID;
  LAUNCH_GEMM(0, EPI_SILU_F16, false, false, p, 4 * N_TOK, HID, 1);

  const long qk_ch = (long)N_TOK * DQK;  // per-role stride in qk_h
  const h16* quad_q = qk_h + 0 * qk_ch;
  const h16* lin_q  = qk_h + 1 * qk_ch;
  const h16* quad_k = qk_h + 2 * qk_ch;
  const h16* lin_k  = qk_h + 3 * qk_ch;

  // lin_k (8192 x 128) -> lin_k^T (128 x 8192)
  {
    long n = (long)N_TOK * DQK;
    flash_tr_h<<<dim3((unsigned)((n + 255) / 256)), 256, 0, stream>>>(lin_k, linkT, N_TOK, DQK);
  }

  // 4) attn_g = mask(relu(qq_g @ qk_g^T / G)^2) : per group, 256x256, K=128
  p = GemmP{};
  p.A = quad_q; p.lda = DQK; p.strideA = (long)G_SZ * DQK;
  p.B = quad_k; p.ldb = DQK; p.strideB = (long)G_SZ * DQK;   // NxK row-major = qk^T
  p.K = DQK; p.scale = 1.f / G_SZ;
  p.Ch = attn_h; p.ldc = G_SZ; p.strideC = (long)G_SZ * G_SZ;
  LAUNCH_GEMM(0, EPI_ATTN_F16, false, false, p, G_SZ, G_SZ, NGRP);

  // 5) kv_g = lk_g^T @ v_g / G : per group, 128x2048, K=256
  p = GemmP{};
  p.A = linkT; p.lda = N_TOK; p.strideA = G_SZ;   // [d][g*256 + k]
  p.B = v0T; p.ldb = N_TOK; p.strideB = G_SZ;     // [e][g*256 + k]
  p.K = G_SZ; p.scale = 1.f / G_SZ;
  p.Cf = kv_f; p.ldc = HID; p.strideC = (long)DQK * HID;
  LAUNCH_GEMM(0, EPI_SCALE_F32, false, false, p, DQK, HID, NGRP);

  // 6) exclusive group cumsum of kv (f32 -> f16, transposed per group)
  {
    long per = (long)DQK * HID;
    flash_cumsum<<<dim3((unsigned)((per + 255) / 256)), 256, 0, stream>>>(kv_f, kvpreT, per);
  }

  // 7) causal depthwise conv of v0T -> convT_f
  {
    long tot = (long)HID * (N_TOK / 8);
    flash_conv<<<dim3((unsigned)((tot + 255) / 256)), 256, 0, stream>>>(v0T, conv_w, convT_f);
  }

  // 8) outpre = attn_g @ v_g + conv : per group 256x2048, K=256
  p = GemmP{};
  p.A = attn_h; p.lda = G_SZ; p.strideA = (long)G_SZ * G_SZ;
  p.B = v0T; p.ldb = N_TOK; p.strideB = G_SZ;
  p.K = G_SZ;
  p.Cf = outpre; p.ldc = HID; p.strideC = (long)G_SZ * HID;
  p.add = convT_f; p.ldadd = N_TOK; p.strideAdd = G_SZ;   // convT[e][g*256+m]
  LAUNCH_GEMM(0, EPI_ADD_F32, false, true, p, G_SZ, HID, NGRP);

  // 9) outpre += lq_g @ kvpre_g : per group 256x2048, K=128
  p = GemmP{};
  p.A = lin_q; p.lda = DQK; p.strideA = (long)G_SZ * DQK;
  p.B = kvpreT; p.ldb = DQK; p.strideB = (long)HID * DQK;  // per-group (2048 x 128)
  p.K = DQK;
  p.Cf = outpre; p.ldc = HID; p.strideC = (long)G_SZ * HID;
  LAUNCH_GEMM(0, EPI_ACC_F32, false, false, p, G_SZ, HID, NGRP);

  // 10) y_c = (gate_c * outpre) @ W_out + b_out + x_c : M=32768, N=1024, K=2048
  p = GemmP{};
  p.A = gate_h; p.lda = HID;
  p.A2 = outpre; p.lda2 = HID; p.arow_mod = N_TOK;
  p.B = WoutT; p.ldb = HID;
  p.K = HID; p.bias = b_out;
  p.Cf = out; p.ldc = DIM;
  p.add = x; p.ldadd = DIM;
  LAUNCH_GEMM(2, EPI_FINAL_F32, false, false, p, 4 * N_TOK, DIM, 1);
}
